// AttentionLayer_13615046328640
// MI455X (gfx1250) — matrix operand order repaired and checked by reading, not compiled, not run on hardware
//
#include <hip/hip_runtime.h>

// ---------------------------------------------------------------------------
// Problem constants (from reference): B=16, C=512, W=H=64 -> S=4096, D=16
// ---------------------------------------------------------------------------
#define NB 16
#define NC 512
#define NS 4096
#define ND 16

typedef __attribute__((ext_vector_type(16))) __bf16 v16bf;
typedef __attribute__((ext_vector_type(8)))  float  v8f;
typedef __attribute__((ext_vector_type(4)))  int    v4i;

#if __has_builtin(__builtin_amdgcn_global_load_async_to_lds_b128) && \
    __has_builtin(__builtin_amdgcn_s_wait_asynccnt)
#define HAVE_ASYNC_LDS 1
#else
#define HAVE_ASYNC_LDS 0
#endif

// ===========================================================================
// Kernel 1: q/k projection.  q[m,d] = sum_s x[m,s]*Wq[d,s] + bq[d]  (m=b*C+c)
// One wave computes a 16x16 tile of q AND of k (shared A fragment).
// K-loop over S=4096 in steps of 32 -> 128 iterations x 2 WMMA.
// A-frag layout (16-bit A 16x32): lane m=l&15, h=l>>4,
//   element e -> K = e + h*16 (same order as B)
// B-frag layout (16-bit B 32x16): lane n=l&15, h=l>>4, element e -> K = e+h*16
// ===========================================================================
__global__ __launch_bounds__(256)
void qk_wmma_kernel(const float* __restrict__ x,
                    const float* __restrict__ Wq, const float* __restrict__ bq,
                    const float* __restrict__ Wk, const float* __restrict__ bk,
                    float* __restrict__ qout, float* __restrict__ kout)
{
    const int lane = threadIdx.x & 31;
    const int wave = threadIdx.x >> 5;
    const int tile = blockIdx.x * 8 + wave;      // 0..511 (8192 rows / 16)
    const int m0   = tile * 16;
    const int ml   = lane & 15;
    const int h    = lane >> 4;

    const float* xrow  = x  + (size_t)(m0 + ml) * NS;
    const float* wqrow = Wq + (size_t)ml * NS;   // B-matrix column n = ml
    const float* wkrow = Wk + (size_t)ml * NS;

    v8f accq = {};
    v8f acck = {};

    for (int s0 = 0; s0 < NS; s0 += 32) {
        union { v16bf v; __bf16 e[16]; } a, bqf, bkf;
        const float* pa1 = xrow + s0 + h * 16;       // K = h*16  + 0..7
        const float* pa2 = xrow + s0 + h * 16 + 8;   // K = h*16+8+ 0..7
        #pragma unroll
        for (int e = 0; e < 8; ++e) {
            a.e[e]     = (__bf16)pa1[e];
            a.e[8 + e] = (__bf16)pa2[e];
        }
        const float* pq = wqrow + s0 + h * 16;       // K = h*16 + e
        const float* pk = wkrow + s0 + h * 16;
        #pragma unroll
        for (int e = 0; e < 16; ++e) {
            bqf.e[e] = (__bf16)pq[e];
            bkf.e[e] = (__bf16)pk[e];
        }
        accq = __builtin_amdgcn_wmma_f32_16x16x32_bf16(
                   false, a.v, false, bqf.v, (short)0, accq, false, false);
        acck = __builtin_amdgcn_wmma_f32_16x16x32_bf16(
                   false, a.v, false, bkf.v, (short)0, acck, false, false);
    }

    // C/D tile: VGPR r -> row r + 8*h, col = ml
    const float biasq = bq[ml];
    const float biask = bk[ml];
    #pragma unroll
    for (int r = 0; r < 8; ++r) {
        const int row = m0 + r + 8 * h;
        qout[row * ND + ml] = accq[r] + biasq;
        kout[row * ND + ml] = acck[r] + biask;
    }
}

// ===========================================================================
// Kernel 2: fused kq + softmax over the batch axis + transpose to bf16.
//   s[b] = dot(q[b,i,:], k[b,j,:])   (16 MACs, q/k are 1MB total -> L2 hot)
//   A[b,i,j] = exp(s[b]-max_b)/sum_b
//   At[b][j][i] = A[b][i][j]   (bf16, exactly the A-matrix layout kernel 3 wants)
// ===========================================================================
__global__ __launch_bounds__(256)
void softmax_bt_kernel(const float* __restrict__ q, const float* __restrict__ k,
                       __bf16* __restrict__ At)
{
    const int i = blockIdx.x * blockDim.x + threadIdx.x;  // 0..511
    const int j = blockIdx.y;                             // 0..511

    float s[NB];
    #pragma unroll
    for (int b = 0; b < NB; ++b) {
        const float* qr = q + ((size_t)(b * NC + i)) * ND;
        const float* kr = k + ((size_t)(b * NC + j)) * ND;
        float acc = 0.f;
        #pragma unroll
        for (int d = 0; d < ND; ++d) acc += qr[d] * kr[d];
        s[b] = acc;
    }
    float mx = s[0];
    #pragma unroll
    for (int b = 1; b < NB; ++b) mx = fmaxf(mx, s[b]);
    float sum = 0.f;
    #pragma unroll
    for (int b = 0; b < NB; ++b) { s[b] = __expf(s[b] - mx); sum += s[b]; }
    const float inv = 1.0f / sum;
    #pragma unroll
    for (int b = 0; b < NB; ++b) {
        At[((size_t)b * NC + j) * NC + i] = (__bf16)(s[b] * inv);
    }
}

// ===========================================================================
// Kernel 3: dominant GEMM, per batch:  out[o,s] = sum_i At[o,i] * x[i,s]
//   M=512 (o), N=4096 (s), K=512 (i).  34.4 GFLOP total in bf16 WMMA.
// Block tile 128x128, 8 waves (4xM, 2xN): wave tile 32x64 = 8 accumulators
//   -> 8 WMMA per k-step against 12 LDS b128 fragment reads.
// At tile: staged via GLOBAL_LOAD_ASYNC_TO_LDS_B128 (ASYNCcnt) when available.
// x tile:  fp32 -> bf16 K-transposed with PACKED ds_store_b64 (4 bf16 along k
//          are contiguous in Bs[n][k]) -- 4 LDS stores/thread/step, not 16.
// ===========================================================================
#define BM 128
#define BN 128
#define BK 32
#define LDA 40
#define LDB 40

__global__ __launch_bounds__(256)
void out_wmma_kernel(const __bf16* __restrict__ At, const float* __restrict__ x,
                     float* __restrict__ out)
{
    __shared__ __bf16 As[BM * LDA];   // [m][k]
    __shared__ __bf16 Bs[BN * LDB];   // [n][k]  (K-transposed x tile)

    const int b    = blockIdx.z;
    const int bm   = blockIdx.y * BM;
    const int bn   = blockIdx.x * BN;
    const int t    = threadIdx.x;
    const int lane = t & 31;
    const int wave = t >> 5;
    const int wm   = (wave >> 1) * 32;   // 0,32,64,96
    const int wn   = (wave & 1) * 64;    // 0,64
    const int ml   = lane & 15;
    const int h    = lane >> 4;

    const __bf16* Ab = At + ((size_t)b * NC + bm) * NC;  // rows o, stride NC
    const float*  Xb = x  + ((size_t)b * NC) * NS;       // rows i, stride NS

    v8f acc[2][4] = {{v8f{}, v8f{}, v8f{}, v8f{}},
                     {v8f{}, v8f{}, v8f{}, v8f{}}};

    // global->LDS work split (256 threads)
    const int arow = t >> 1;             // 128 A-rows, 2 threads/row
    const int acg  = (t & 1) * 16;       // 16 bf16 (2x16B) each
    const int kq4  = (t >> 5) * 4;       // 8 k-quads of 4 rows
    const int cs4  = (t & 31) * 4;       // 32 col-segments of 4 columns

    for (int k0 = 0; k0 < NC; k0 += BK) {
        __syncthreads();

        // ---- Stage At tile (bf16, row-major) ----
#if HAVE_ASYNC_LDS
        {
            const __bf16* g = Ab + (size_t)arow * NC + k0 + acg;
            __builtin_amdgcn_global_load_async_to_lds_b128(
                (v4i*)g,       (v4i*)&As[arow * LDA + acg],     0, 0);
            __builtin_amdgcn_global_load_async_to_lds_b128(
                (v4i*)(g + 8), (v4i*)&As[arow * LDA + acg + 8], 0, 0);
        }
#else
        {
            const uint4* src = (const uint4*)(Ab + (size_t)arow * NC + k0 + acg);
            *(uint4*)(&As[arow * LDA + acg])     = src[0];
            *(uint4*)(&As[arow * LDA + acg + 8]) = src[1];
        }
#endif

        // ---- Stage x tile: fp32 -> bf16, K-transposed, packed b64 stores ----
        {
            // 4-row x 4-col micro-tile per thread
            const float* r0 = Xb + (size_t)(k0 + kq4 + 0) * NS + bn + cs4;
            const float* r1 = Xb + (size_t)(k0 + kq4 + 1) * NS + bn + cs4;
            const float* r2 = Xb + (size_t)(k0 + kq4 + 2) * NS + bn + cs4;
            const float* r3 = Xb + (size_t)(k0 + kq4 + 3) * NS + bn + cs4;
            float4 f0 = *(const float4*)r0;
            float4 f1 = *(const float4*)r1;
            float4 f2 = *(const float4*)r2;
            float4 f3 = *(const float4*)r3;
            const float c0[4] = {f0.x, f1.x, f2.x, f3.x};
            const float c1[4] = {f0.y, f1.y, f2.y, f3.y};
            const float c2[4] = {f0.z, f1.z, f2.z, f3.z};
            const float c3[4] = {f0.w, f1.w, f2.w, f3.w};
            const float* cols[4] = {c0, c1, c2, c3};
            #pragma unroll
            for (int c = 0; c < 4; ++c) {
                union { uint2 u; __bf16 e[4]; } pk;
                #pragma unroll
                for (int r = 0; r < 4; ++r) pk.e[r] = (__bf16)cols[c][r];
                *(uint2*)(&Bs[(cs4 + c) * LDB + kq4]) = pk.u;
            }
        }

        // Prefetch next k-step's x rows (global_prefetch_b8, no LOADcnt cost)
        if (k0 + BK < NC) {
            __builtin_prefetch(Xb + (size_t)(k0 + BK + kq4) * NS + bn + cs4, 0, 3);
        }

#if HAVE_ASYNC_LDS
        __builtin_amdgcn_s_wait_asynccnt(0);
#endif
        __syncthreads();

        // A fragments (2 M-subtiles): element e -> K = e + h*16
        union { v16bf v; uint4 q[2]; } a[2];
        #pragma unroll
        for (int mm = 0; mm < 2; ++mm) {
            const int m = wm + mm * 16 + ml;
            a[mm].q[0] = *(const uint4*)(&As[m * LDA + h * 16]);
            a[mm].q[1] = *(const uint4*)(&As[m * LDA + h * 16 + 8]);
        }

        #pragma unroll
        for (int tt = 0; tt < 4; ++tt) {
            // B fragment: lane col n, element e -> K = e + h*16 (two 16B reads)
            union { v16bf v; uint4 q[2]; } bf;
            const int n = wn + tt * 16 + ml;
            bf.q[0] = *(const uint4*)(&Bs[n * LDB + h * 16]);
            bf.q[1] = *(const uint4*)(&Bs[n * LDB + h * 16 + 8]);
            acc[0][tt] = __builtin_amdgcn_wmma_f32_16x16x32_bf16(
                             false, a[0].v, false, bf.v, (short)0, acc[0][tt], false, false);
            acc[1][tt] = __builtin_amdgcn_wmma_f32_16x16x32_bf16(
                             false, a[1].v, false, bf.v, (short)0, acc[1][tt], false, false);
        }
    }

    // Epilogue: C/D layout -> out[b][o][s]
    #pragma unroll
    for (int mm = 0; mm < 2; ++mm) {
        #pragma unroll
        for (int tt = 0; tt < 4; ++tt) {
            const int scol = bn + wn + tt * 16 + ml;
            #pragma unroll
            for (int r = 0; r < 8; ++r) {
                const int orow = bm + wm + mm * 16 + r + 8 * h;
                out[((size_t)b * NC + orow) * NS + scol] = acc[mm][tt][r];
            }
        }
    }
}

// ===========================================================================
// Host-side launcher
// ===========================================================================
extern "C" void kernel_launch(void* const* d_in, const int* in_sizes, int n_in,
                              void* d_out, int out_size, void* d_ws, size_t ws_size,
                              hipStream_t stream)
{
    const float* x  = (const float*)d_in[0];
    const float* Wq = (const float*)d_in[1];
    const float* bq = (const float*)d_in[2];
    const float* Wk = (const float*)d_in[3];
    const float* bk = (const float*)d_in[4];
    float* out = (float*)d_out;

    char* ws = (char*)d_ws;
    float*  qbuf = (float*)(ws);                       // 8192*16*4  = 512 KB
    float*  kbuf = (float*)(ws + (512u << 10));        // 512 KB
    __bf16* At   = (__bf16*)(ws + (1u << 20));         // 16*512*512*2 = 8 MB

    // K1: 512 wave-tiles of 16 rows, 8 waves/block -> 64 blocks
    qk_wmma_kernel<<<64, 256, 0, stream>>>(x, Wq, bq, Wk, bk, qbuf, kbuf);

    // K2: grid (i-tiles, j) = (2, 512), 256 threads
    softmax_bt_kernel<<<dim3(2, 512), 256, 0, stream>>>(qbuf, kbuf, At);

    // K3: grid (N/BN, M/BM, B) = (32, 4, 16), 256 threads
    out_wmma_kernel<<<dim3(NS / BN, NC / BM, NB), 256, 0, stream>>>(At, x, out);
}
